// SAtt_61959198212131
// MI455X (gfx1250) — matrix operand order repaired and checked by reading, not compiled, not run on hardware
//
#include <hip/hip_runtime.h>

#define B_ 8
#define C_ 256
#define N_ 2304           // 48*48
#define NBLK 36           // N_/64
#define EPS_ 1e-5f

typedef __attribute__((ext_vector_type(16))) _Float16 v16h;
typedef __attribute__((ext_vector_type(8)))  _Float16 v8h;
typedef __attribute__((ext_vector_type(8)))  float    v8f;

__device__ __forceinline__ v8f wmma16(v16h a, v16h b, v8f c) {
  return __builtin_amdgcn_wmma_f32_16x16x32_f16(
      /*neg_a=*/false, a, /*neg_b=*/false, b,
      /*c_mod=*/(short)0, c, /*reuse_a=*/false, /*reuse_b=*/false);
}

// A fragment (16x32, f16) from row-major [rows][ld] matrix, tile at (r0, k0).
// Lane L: M = L%16; K = k0 + (L/16)*16 + e.
__device__ __forceinline__ v16h frag_a(const _Float16* __restrict__ p, int ld,
                                       int r0, int k0, int lane) {
  const int m = lane & 15, hi = lane >> 4;
  const _Float16* base = p + (size_t)(r0 + m) * ld + k0 + hi * 16;
  v8h lo = *(const v8h*)(base);
  v8h hi8 = *(const v8h*)(base + 8);
  v16h out;
#pragma unroll
  for (int i = 0; i < 8; ++i) { out[i] = lo[i]; out[i + 8] = hi8[i]; }
  return out;
}

// B fragment (32x16) loaded from B^T stored row-major [ncols][ld] (K contiguous).
// Lane L: N = n0 + L%16; K = k0 + (L/16)*16 + e.
__device__ __forceinline__ v16h frag_bt(const _Float16* __restrict__ p, int ld,
                                        int n0, int k0, int lane) {
  const int n = lane & 15, kh = lane >> 4;
  const _Float16* base = p + (size_t)(n0 + n) * ld + k0 + kh * 16;
  v8h a = *(const v8h*)(base);
  v8h b = *(const v8h*)(base + 8);
  v16h out;
#pragma unroll
  for (int i = 0; i < 8; ++i) { out[i] = a[i]; out[i + 8] = b[i]; }
  return out;
}

// ---------------- k0: convert the four CxC weights to f16 -------------------
__global__ __launch_bounds__(256) void k0_cvt(const float* __restrict__ Wq,
                                              const float* __restrict__ Wk,
                                              const float* __restrict__ Wv,
                                              const float* __restrict__ Wz,
                                              _Float16* __restrict__ Wh) {
  int i = blockIdx.x * 256 + threadIdx.x;      // 0 .. 4*C*C
  int w = i >> 16, j = i & 65535;              // C*C == 65536
  float v;
  if (w == 0)      v = Wq[j];
  else if (w == 1) v = Wk[j];
  else if (w == 2) v = Wv[j];
  else             v = Wz[j];
  Wh[i] = (_Float16)v;
}

// ---------------- k1: Q/K/V projections (Wp @ X) ----------------------------
// grid: (B_*NBLK, 3). block: 256 (8 waves). Out tile: 256 o x 64 n.
// p==0: Q stored transposed f16 [B][N][C]; p==1/2: K/V stored f16 [B][C][N].
__global__ __launch_bounds__(256) void k1_proj(const float* __restrict__ Xq,
                                               const float* __restrict__ Xkv,
                                               const _Float16* __restrict__ Wh,
                                               _Float16* __restrict__ Qt,
                                               _Float16* __restrict__ Kh,
                                               _Float16* __restrict__ Vh) {
  __shared__ __align__(16) _Float16 lds_x[64 * 32];   // [n][c] transposed tile
  const int p = blockIdx.y;
  const int b = blockIdx.x / NBLK;
  const int nb = (blockIdx.x % NBLK) * 64;
  const float* __restrict__ Xb = ((p == 0) ? Xq : Xkv) + (size_t)b * C_ * N_;
  const _Float16* __restrict__ W = Wh + (size_t)p * C_ * C_;
  const int t = threadIdx.x, lane = t & 31, wave = t >> 5;
  const int ot0 = wave * 2;                 // two o-tiles per wave
  v8f acc[2][4] = {};

  const int cl = t >> 3;                    // 0..31 : c within k-step
  const int n8 = (t & 7) * 8;               // 0..56 : 8-wide n chunk

  for (int s = 0; s < C_; s += 32) {
    // stage X^T tile [64 n][32 c] into LDS with f32->f16 convert
    const float4* src = (const float4*)(Xb + (size_t)(s + cl) * N_ + nb + n8);
    float4 u0 = src[0], u1 = src[1];
    if (s + 32 < C_)
      __builtin_prefetch(Xb + (size_t)(s + 32 + cl) * N_ + nb + n8, 0, 1);
    lds_x[(n8 + 0) * 32 + cl] = (_Float16)u0.x;
    lds_x[(n8 + 1) * 32 + cl] = (_Float16)u0.y;
    lds_x[(n8 + 2) * 32 + cl] = (_Float16)u0.z;
    lds_x[(n8 + 3) * 32 + cl] = (_Float16)u0.w;
    lds_x[(n8 + 4) * 32 + cl] = (_Float16)u1.x;
    lds_x[(n8 + 5) * 32 + cl] = (_Float16)u1.y;
    lds_x[(n8 + 6) * 32 + cl] = (_Float16)u1.z;
    lds_x[(n8 + 7) * 32 + cl] = (_Float16)u1.w;
    __syncthreads();

    v16h a0 = frag_a(W, C_, (ot0 + 0) * 16, s, lane);
    v16h a1 = frag_a(W, C_, (ot0 + 1) * 16, s, lane);
#pragma unroll
    for (int nt = 0; nt < 4; ++nt) {
      const int nn = lane & 15, kh = lane >> 4;
      const _Float16* bp = &lds_x[(nt * 16 + nn) * 32 + kh * 16];
      v8h blo = *(const v8h*)(bp);
      v8h bhi = *(const v8h*)(bp + 8);
      v16h bf;
#pragma unroll
      for (int i = 0; i < 8; ++i) { bf[i] = blo[i]; bf[i + 8] = bhi[i]; }
      acc[0][nt] = wmma16(a0, bf, acc[0][nt]);
      acc[1][nt] = wmma16(a1, bf, acc[1][nt]);
    }
    __syncthreads();
  }

  if (p == 0) {
    _Float16* __restrict__ Qtb = Qt + (size_t)b * N_ * C_;
#pragma unroll
    for (int i = 0; i < 2; ++i)
#pragma unroll
      for (int nt = 0; nt < 4; ++nt) {
        int n = nb + nt * 16 + (lane & 15);
        int o0 = (ot0 + i) * 16 + (lane >> 4) * 8;
        v8h h;
#pragma unroll
        for (int r = 0; r < 8; ++r) h[r] = (_Float16)acc[i][nt][r];
        *(v8h*)(Qtb + (size_t)n * C_ + o0) = h;
      }
  } else {
    _Float16* __restrict__ Ob = ((p == 1) ? Kh : Vh) + (size_t)b * C_ * N_;
#pragma unroll
    for (int i = 0; i < 2; ++i)
#pragma unroll
      for (int nt = 0; nt < 4; ++nt) {
        int n = nb + nt * 16 + (lane & 15);
        int c0 = (ot0 + i) * 16 + (lane >> 4) * 8;
#pragma unroll
        for (int r = 0; r < 8; ++r)
          Ob[(size_t)(c0 + r) * N_ + n] = (_Float16)acc[i][nt][r];
      }
  }
}

// ---------------- k2: KV[b] = K V^T / N (CxC) -------------------------------
// grid: (B_, 16). block 256 (8 waves). block tile 64c x 64d, 2 tiles/wave.
__global__ __launch_bounds__(256) void k2_kv(const _Float16* __restrict__ Kh,
                                             const _Float16* __restrict__ Vh,
                                             _Float16* __restrict__ KVh) {
  const int b = blockIdx.x, tile = blockIdx.y;
  const int cb = (tile >> 2) * 64, db = (tile & 3) * 64;
  const int t = threadIdx.x, lane = t & 31, wave = t >> 5;
  const int ct = wave & 3, dt0 = (wave >> 2) * 2;
  const _Float16* __restrict__ Kb = Kh + (size_t)b * C_ * N_;
  const _Float16* __restrict__ Vb = Vh + (size_t)b * C_ * N_;
  v8f acc[2] = {};
  for (int n0 = 0; n0 < N_; n0 += 32) {
    v16h a  = frag_a (Kb, N_, cb + ct * 16, n0, lane);
    v16h b0 = frag_bt(Vb, N_, db + (dt0 + 0) * 16, n0, lane);
    v16h b1 = frag_bt(Vb, N_, db + (dt0 + 1) * 16, n0, lane);
    acc[0] = wmma16(a, b0, acc[0]);
    acc[1] = wmma16(a, b1, acc[1]);
  }
  const float sc = 1.0f / (float)N_;
  _Float16* __restrict__ KVb = KVh + (size_t)b * C_ * C_;
#pragma unroll
  for (int j = 0; j < 2; ++j) {
    int d = db + (dt0 + j) * 16 + (lane & 15);
    int c0 = cb + ct * 16 + (lane >> 4) * 8;
#pragma unroll
    for (int r = 0; r < 8; ++r)
      KVb[(size_t)(c0 + r) * C_ + d] = (_Float16)(acc[j][r] * sc);
  }
}

// ---------------- k3: M[b] = inv * (Wz @ KV^T) ------------------------------
// grid: (B_, 16). Same wave layout as k2. M stored f16 [B][o][c].
__global__ __launch_bounds__(256) void k3_m(const _Float16* __restrict__ Wzh,
                                            const _Float16* __restrict__ KVh,
                                            const float* __restrict__ gamma,
                                            const float* __restrict__ var,
                                            _Float16* __restrict__ Mh) {
  const int b = blockIdx.x, tile = blockIdx.y;
  const int ob = (tile >> 2) * 64, cb = (tile & 3) * 64;
  const int t = threadIdx.x, lane = t & 31, wave = t >> 5;
  const int ot = wave & 3, ct0 = (wave >> 2) * 2;
  const _Float16* __restrict__ KVb = KVh + (size_t)b * C_ * C_;
  v8f acc[2] = {};
  for (int d0 = 0; d0 < C_; d0 += 32) {
    v16h a  = frag_a (Wzh, C_, ob + ot * 16, d0, lane);
    v16h b0 = frag_bt(KVb, C_, cb + (ct0 + 0) * 16, d0, lane);
    v16h b1 = frag_bt(KVb, C_, cb + (ct0 + 1) * 16, d0, lane);
    acc[0] = wmma16(a, b0, acc[0]);
    acc[1] = wmma16(a, b1, acc[1]);
  }
  _Float16* __restrict__ Mb = Mh + (size_t)b * C_ * C_;
#pragma unroll
  for (int j = 0; j < 2; ++j) {
    int c = cb + (ct0 + j) * 16 + (lane & 15);
    int o0 = ob + ot * 16 + (lane >> 4) * 8;
#pragma unroll
    for (int r = 0; r < 8; ++r) {
      int o = o0 + r;
      float inv = gamma[o] * __frsqrt_rn(var[o] + EPS_);
      Mb[(size_t)o * C_ + c] = (_Float16)(acc[j][r] * inv);
    }
  }
}

// ---------------- k4: out = M @ Q + shift + residual ------------------------
// grid: B_*NBLK. block 256 (8 waves). Out tile 256 o x 64 n.
__global__ __launch_bounds__(256) void k4_out(const _Float16* __restrict__ Mh,
                                              const _Float16* __restrict__ Qt,
                                              const float* __restrict__ Xq,
                                              const float* __restrict__ gamma,
                                              const float* __restrict__ beta,
                                              const float* __restrict__ mean,
                                              const float* __restrict__ var,
                                              float* __restrict__ out) {
  const int b = blockIdx.x / NBLK;
  const int nb = (blockIdx.x % NBLK) * 64;
  const int t = threadIdx.x, lane = t & 31, wave = t >> 5;
  const int ot0 = wave * 2;
  const _Float16* __restrict__ Mb  = Mh + (size_t)b * C_ * C_;
  const _Float16* __restrict__ Qtb = Qt + (size_t)b * N_ * C_;
  v8f acc[2][4] = {};
  for (int c0 = 0; c0 < C_; c0 += 32) {
    v16h a0 = frag_a(Mb, C_, (ot0 + 0) * 16, c0, lane);
    v16h a1 = frag_a(Mb, C_, (ot0 + 1) * 16, c0, lane);
#pragma unroll
    for (int nt = 0; nt < 4; ++nt) {
      v16h bf = frag_bt(Qtb, C_, nb + nt * 16, c0, lane);
      acc[0][nt] = wmma16(a0, bf, acc[0][nt]);
      acc[1][nt] = wmma16(a1, bf, acc[1][nt]);
    }
  }
  const float* __restrict__ Xb = Xq + (size_t)b * C_ * N_;
  float* __restrict__ Ob = out + (size_t)b * C_ * N_;
#pragma unroll
  for (int i = 0; i < 2; ++i)
#pragma unroll
    for (int nt = 0; nt < 4; ++nt) {
      int n = nb + nt * 16 + (lane & 15);
      int o0 = (ot0 + i) * 16 + (lane >> 4) * 8;
#pragma unroll
      for (int r = 0; r < 8; ++r) {
        int o = o0 + r;
        float inv = gamma[o] * __frsqrt_rn(var[o] + EPS_);
        float sh = beta[o] - mean[o] * inv;
        Ob[(size_t)o * N_ + n] = acc[i][nt][r] + sh + Xb[(size_t)o * N_ + n];
      }
    }
}

extern "C" void kernel_launch(void* const* d_in, const int* in_sizes, int n_in,
                              void* d_out, int out_size, void* d_ws, size_t ws_size,
                              hipStream_t stream) {
  const float* Xq    = (const float*)d_in[0];
  const float* Xkv   = (const float*)d_in[1];
  const float* Wq    = (const float*)d_in[2];
  const float* Wk    = (const float*)d_in[3];
  const float* Wv    = (const float*)d_in[4];
  const float* Wz    = (const float*)d_in[5];
  const float* gamma = (const float*)d_in[6];
  const float* beta  = (const float*)d_in[7];
  const float* mean  = (const float*)d_in[8];
  const float* var   = (const float*)d_in[9];
  float* out = (float*)d_out;

  char* ws = (char*)d_ws;
  size_t off = 0;
  auto carve = [&](size_t bytes) {
    void* p = ws + off;
    off += (bytes + 255) & ~(size_t)255;
    return p;
  };
  _Float16* Wh  = (_Float16*)carve(4 * (size_t)C_ * C_ * 2);       // Wq|Wk|Wv|Wz f16
  _Float16* Qt  = (_Float16*)carve((size_t)B_ * N_ * C_ * 2);      // Q^T  [B][N][C]
  _Float16* Kh  = (_Float16*)carve((size_t)B_ * C_ * N_ * 2);      // K    [B][C][N]
  _Float16* Vh  = (_Float16*)carve((size_t)B_ * C_ * N_ * 2);      // V    [B][C][N]
  _Float16* KVh = (_Float16*)carve((size_t)B_ * C_ * C_ * 2);      // KV/N [B][C][C]
  _Float16* Mh  = (_Float16*)carve((size_t)B_ * C_ * C_ * 2);      // M    [B][C][C]

  k0_cvt<<<(4 * C_ * C_) / 256, 256, 0, stream>>>(Wq, Wk, Wv, Wz, Wh);
  k1_proj<<<dim3(B_ * NBLK, 3), 256, 0, stream>>>(Xq, Xkv, Wh, Qt, Kh, Vh);
  k2_kv<<<dim3(B_, 16), 256, 0, stream>>>(Kh, Vh, KVh);
  k3_m<<<dim3(B_, 16), 256, 0, stream>>>(Wh + 3 * (size_t)C_ * C_, KVh, gamma, var, Mh);
  k4_out<<<B_ * NBLK, 256, 0, stream>>>(Mh, Qt, Xq, gamma, beta, mean, var, out);
}
